// RelationLayer_22582938043065
// MI455X (gfx1250) — compile-verified, matrix operand order checked by reading, not run on hardware
//
#include <hip/hip_runtime.h>

typedef __attribute__((ext_vector_type(16))) __bf16          v16bf;
typedef __attribute__((ext_vector_type(16))) unsigned short  v16u;
typedef __attribute__((ext_vector_type(8)))  unsigned short  v8u;
typedef __attribute__((ext_vector_type(8)))  float           v8f;

#define NEG_SLOPE 0.1f
#define BN_EPS    1e-5f

__device__ __forceinline__ float lrelu(float z){ return z > 0.f ? z : NEG_SLOPE * z; }

__device__ __forceinline__ unsigned short f2bf(float f){
  unsigned int u = __float_as_uint(f);
  u += 0x7FFFu + ((u >> 16) & 1u);          // round-to-nearest-even
  return (unsigned short)(u >> 16);
}

// K-index a lane-half holds for fragment element e (16-bit A/B operand layout, ISA 7.12.2)
__device__ __forceinline__ int kmap(int e, int h){ return (e < 8) ? (e + 8*h) : (e + 8 + 8*h); }

__device__ __forceinline__ v8f wmma_bf16(v16u a, v16u b, v8f c){
  return __builtin_amdgcn_wmma_f32_16x16x32_bf16(
      false, __builtin_bit_cast(v16bf, a),
      false, __builtin_bit_cast(v16bf, b),
      (short)0, c, false, false);
}

__device__ __forceinline__ v16u ld_frag(const unsigned short* p0, const unsigned short* p1){
  v8u lo = *(const v8u*)p0;
  v8u hi = *(const v8u*)p1;
  return __builtin_shufflevector(lo, hi, 0,1,2,3,4,5,6,7,8,9,10,11,12,13,14,15);
}

// ---------------------------------------------------------------- utilities
__global__ void k_zero(float* p, int n){
  int i = blockIdx.x*blockDim.x + threadIdx.x;
  if (i < n) p[i] = 0.f;
}

// x: (L=100,B=32,E=512) f32  ->  (B,E,L) bf16
__global__ void k_prep_in0(const float* __restrict__ x, unsigned short* __restrict__ out){
  int i = blockIdx.x*blockDim.x + threadIdx.x;
  if (i >= 32*512*100) return;
  int l = i % 100; int e = (i/100) % 512; int b = i/(100*512);
  out[i] = f2bf(x[(l*32 + b)*512 + e]);
}

// conv weights (F=256, Ktot=nkt*32) row-major f32 -> WMMA A-fragment layout [mt][kt][lane][16] bf16
__global__ void k_prep_afrag(const float* __restrict__ w, unsigned short* __restrict__ out, int nkt){
  int i = blockIdx.x*blockDim.x + threadIdx.x;
  if (i >= 16*nkt*512) return;
  int e = i & 15; int lane = (i>>4)&31; int t = i>>9;
  int kt = t % nkt; int mt = t / nkt;
  int f = mt*16 + (lane&15); int h = lane>>4;
  int k = kt*32 + kmap(e,h);
  out[i] = f2bf(w[(size_t)f*(nkt*32) + k]);
}

// weights (K x 128) row-major f32 -> WMMA B-fragment layout [kt][nt(8)][lane][16] bf16 (zero pad k>=klimit)
__global__ void k_prep_bfrag(const float* __restrict__ w, unsigned short* __restrict__ out,
                             int nkt, int klimit){
  int i = blockIdx.x*blockDim.x + threadIdx.x;
  if (i >= nkt*8*512) return;
  int e = i & 15; int lane = (i>>4)&31; int t = i>>9;
  int nt = t & 7; int kt = t >> 3;
  int n = nt*16 + (lane&15); int h = lane>>4;
  int k = kt*32 + kmap(e,h);
  out[i] = (k < klimit) ? f2bf(w[(size_t)k*128 + n]) : (unsigned short)0;
}

// ---------------------------------------------------------------- conv1d(K=3,pad=1) + bias + lrelu + BN stats
// GEMM: M=F(=256, 16 tiles) x N=16 output positions x K=Cin*3.  4 waves/block, one mt per wave.
__global__ __launch_bounds__(128) void k_conv_wmma(
    const unsigned short* __restrict__ in, const unsigned short* __restrict__ wfrag,
    const float* __restrict__ cb, float* __restrict__ act, float* __restrict__ stats, int Cin){
  __shared__ unsigned short stg[512*18];
  int b = blockIdx.x, lt = blockIdx.y, mg = blockIdx.z;
  int tid = threadIdx.x; int lane = tid & 31; int wv = tid >> 5;
  int nkt = (Cin*3) >> 5;
  int lbase = lt*16;
  // stage input patch [Cin][18] covering l in [lbase-1, lbase+16], zero padded
  int total = Cin*18;
  for (int i = tid; i < total; i += 128){
    int e = i/18, col = i - e*18;
    int lg = lbase - 1 + col;
    stg[i] = (lg >= 0 && lg < 100) ? in[(size_t)(b*Cin + e)*100 + lg] : (unsigned short)0;
  }
  __syncthreads();

  int mt = mg*4 + wv;
  int h = lane >> 4; int nl = lane & 15;
  v8f acc = {0.f,0.f,0.f,0.f,0.f,0.f,0.f,0.f};
  for (int kt = 0; kt < nkt; ++kt){
    const unsigned short* ap = wfrag + ((size_t)(mt*nkt + kt) << 9) + lane*16;
    v16u a = ld_frag(ap, ap + 8);
    v16u bf;
    #pragma unroll
    for (int e = 0; e < 16; ++e){
      int kidx = kt*32 + kmap(e,h);
      int ech = kidx/3, koff = kidx - ech*3;
      bf[e] = stg[ech*18 + nl + koff];
    }
    acc = wmma_bf16(a, bf, acc);
  }
  int lpos = lbase + nl;
  bool lv = lpos < 100;
  #pragma unroll
  for (int r = 0; r < 8; ++r){
    int f = mt*16 + r + 8*h;
    float y = lrelu(acc[r] + cb[f]);
    if (lv) act[(size_t)(b*256 + f)*100 + lpos] = y;
    float v1 = lv ? y : 0.f;
    float v2 = v1*v1;
    #pragma unroll
    for (int m = 1; m < 16; m <<= 1){
      v1 += __shfl_xor(v1, m, 32);
      v2 += __shfl_xor(v2, m, 32);
    }
    if (nl == 0){
      atomicAdd(&stats[f], v1);
      atomicAdd(&stats[256 + f], v2);
    }
  }
}

// BN (training stats over B*L=3200) -> bf16 (B,256,L)
__global__ void k_bn(const float* __restrict__ act, const float* __restrict__ stats,
                     const float* __restrict__ g, const float* __restrict__ be,
                     unsigned short* __restrict__ out){
  int i = blockIdx.x*blockDim.x + threadIdx.x;
  if (i >= 32*256*100) return;
  int f = (i/100) & 255;
  float mean = stats[f] * (1.f/3200.f);
  float var  = stats[256+f] * (1.f/3200.f) - mean*mean;
  float inv  = rsqrtf(var + BN_EPS);
  out[i] = f2bf(g[f]*(act[i]-mean)*inv + be[f]);
}

// xc bf16 (B,256,L) -> xf bf16 (3200 rows x 288) with coord features at d=256,257; zeros 258..287
__global__ void k_prep_xf(const unsigned short* __restrict__ xc, unsigned short* __restrict__ xf){
  int i = blockIdx.x*blockDim.x + threadIdx.x;
  if (i >= 3200*288) return;
  int d = i % 288; int row = i / 288; int l = row % 100; int b = row / 100;
  unsigned short v;
  if (d < 256)       v = xc[(size_t)(b*256 + d)*100 + l];
  else if (d == 256) v = f2bf(((float)l * 0.1f - 2.f) * 0.5f);
  else if (d == 257) v = f2bf((fmodf((float)l, 10.f) - 2.f) * 0.5f);
  else               v = 0;
  xf[i] = v;
}

// hb[b][m] = b1[m] + h[b,:] @ Wc  (Wc = W1 rows 516..1027)
__global__ void k_hb(const float* __restrict__ h, const float* __restrict__ W1,
                     const float* __restrict__ b1, float* __restrict__ hb){
  int i = blockIdx.x*blockDim.x + threadIdx.x;
  if (i >= 32*128) return;
  int m = i & 127, b = i >> 7;
  float acc = b1[m];
  for (int k = 0; k < 512; ++k) acc += h[b*512 + k] * W1[(size_t)(516 + k)*128 + m];
  hb[i] = acc;
}

// out(3200x128) = xf(3200x288) @ W(288x128 fragment-packed) [+ hb[row/100]]
__global__ __launch_bounds__(256) void k_gemm_xf(
    const unsigned short* __restrict__ xf, const unsigned short* __restrict__ wfrag,
    const float* __restrict__ hb, float* __restrict__ out){
  int mt = blockIdx.x; int lane = threadIdx.x & 31; int nt = threadIdx.x >> 5;
  int h = lane >> 4, nl = lane & 15;
  int n = nt*16 + nl;
  v8f acc;
  #pragma unroll
  for (int r = 0; r < 8; ++r){
    int row = mt*16 + r + 8*h;
    acc[r] = hb ? hb[(row/100)*128 + n] : 0.f;
  }
  int arow = mt*16 + nl;
  #pragma unroll
  for (int kt = 0; kt < 9; ++kt){
    const unsigned short* ap = xf + (size_t)arow*288 + kt*32 + 8*h;
    v16u a = ld_frag(ap, ap + 16);
    const unsigned short* wp = wfrag + ((size_t)(kt*8 + nt) << 9) + lane*16;
    v16u w = ld_frag(wp, wp + 8);
    acc = wmma_bf16(a, w, acc);
  }
  #pragma unroll
  for (int r = 0; r < 8; ++r)
    out[(size_t)(mt*16 + r + 8*h)*128 + n] = acc[r];
}

// ---------------------------------------------------------------- fused pairwise relation MLP + reduction
// Two i-rows per task: each weight fragment load feeds TWO v_wmma issues (halves weight traffic/FLOP).
// per tile: g0 = lrelu(C[b,i,:]+A[b,j,:]); 3x (16x128)@(128x128) bf16 WMMA; masked row-sum.
__global__ __launch_bounds__(256) void k_pairwise(
    const float* __restrict__ A, const float* __restrict__ C,
    const unsigned short* __restrict__ wrel,
    const float* __restrict__ b2, const float* __restrict__ b3, const float* __restrict__ b4,
    float* __restrict__ s){
  __shared__ unsigned short actlds[8][2][2048];   // per-wave 2x (16x128) bf16 re-layout tiles
  int b = blockIdx.x, ibase = blockIdx.y * 8;
  int lane = threadIdx.x & 31, wv = threadIdx.x >> 5;
  int h = lane >> 4, nl = lane & 15;
  unsigned short* my0 = actlds[wv][0];
  unsigned short* my1 = actlds[wv][1];
  float sreg[8];
  #pragma unroll
  for (int q = 0; q < 8; ++q) sreg[q] = 0.f;
  const float* biases[3] = {b2, b3, b4};

  for (int t = wv; t < 28; t += 8){           // 4 i-pairs x 7 j-tiles
    int p  = t / 7;
    int i0 = ibase + 2*p;
    if (i0 >= 100) continue;                  // wave-uniform skip of padded i-range
    int jbase = (t - p*7) * 16;
    size_t rowj  = (size_t)(b*100 + jbase + nl) * 128;
    size_t rowi0 = (size_t)(b*100 + i0) * 128;
    size_t rowi1 = rowi0 + 128;

    // prefetch next task's A tile (global_prefetch_b8 path)
    int t2 = t + 8;
    if (t2 < 28){
      int p2 = t2 / 7;
      int i02 = ibase + 2*p2;
      if (i02 < 100){
        size_t rowj2 = (size_t)(b*100 + (t2 - p2*7)*16 + nl) * 128;
        __builtin_prefetch(A + rowj2, 0, 3);
      }
    }

    v16u g0[4], g1[4];
    #pragma unroll
    for (int kt = 0; kt < 4; ++kt){
      int k0 = kt*32 + 8*h;
      v8f aj0 = *(const v8f*)(A + rowj + k0);
      v8f aj1 = *(const v8f*)(A + rowj + k0 + 16);
      v8f ci0 = *(const v8f*)(C + rowi0 + k0);
      v8f ci1 = *(const v8f*)(C + rowi0 + k0 + 16);
      v8f di0 = *(const v8f*)(C + rowi1 + k0);
      v8f di1 = *(const v8f*)(C + rowi1 + k0 + 16);
      v16u u0, u1;
      #pragma unroll
      for (int e = 0; e < 8; ++e){
        u0[e]   = f2bf(lrelu(aj0[e] + ci0[e]));
        u0[e+8] = f2bf(lrelu(aj1[e] + ci1[e]));
        u1[e]   = f2bf(lrelu(aj0[e] + di0[e]));
        u1[e+8] = f2bf(lrelu(aj1[e] + di1[e]));
      }
      g0[kt] = u0; g1[kt] = u1;
    }

    #pragma unroll
    for (int layer = 0; layer < 3; ++layer){
      const unsigned short* wl = wrel + layer*16384;
      const float* bias = biases[layer];
      #pragma unroll
      for (int nt = 0; nt < 8; ++nt){
        float bv = bias[nt*16 + nl];
        v8f acc0 = {bv,bv,bv,bv,bv,bv,bv,bv};
        v8f acc1 = {bv,bv,bv,bv,bv,bv,bv,bv};
        #pragma unroll
        for (int kt = 0; kt < 4; ++kt){
          const unsigned short* wp = wl + ((kt*8 + nt) << 9) + lane*16;
          v16u w = ld_frag(wp, wp + 8);
          acc0 = wmma_bf16(g0[kt], w, acc0);     // one weight fragment, two WMMAs
          acc1 = wmma_bf16(g1[kt], w, acc1);
        }
        #pragma unroll
        for (int r = 0; r < 8; ++r){ acc0[r] = lrelu(acc0[r]); acc1[r] = lrelu(acc1[r]); }
        if (layer < 2){
          #pragma unroll
          for (int r = 0; r < 8; ++r){
            int idx = (r + 8*h)*128 + nt*16 + nl;
            my0[idx] = f2bf(acc0[r]);
            my1[idx] = f2bf(acc1[r]);
          }
        } else {
          #pragma unroll
          for (int r = 0; r < 8; ++r){
            int j = jbase + r + 8*h;
            if (j < 100) sreg[nt] += acc0[r] + acc1[r];
          }
        }
      }
      if (layer < 2){
        #pragma unroll
        for (int kt = 0; kt < 4; ++kt){
          int off = nl*128 + kt*32 + 8*h;
          g0[kt] = ld_frag(my0 + off, my0 + off + 16);  // same-wave DS ordering: no barrier
          g1[kt] = ld_frag(my1 + off, my1 + off + 16);
        }
      }
    }
  }
  #pragma unroll
  for (int nt = 0; nt < 8; ++nt)
    atomicAdd(&s[b*128 + nt*16 + nl], sreg[nt]);
}

// head: out = lrelu(lrelu(s@F1+fb1)@F2+fb2)
__global__ __launch_bounds__(512) void k_head(
    const float* __restrict__ s, const float* __restrict__ F1, const float* __restrict__ fb1,
    const float* __restrict__ F2, const float* __restrict__ fb2, float* __restrict__ out){
  __shared__ float sv[128], tv[128];
  int b = blockIdx.x, tid = threadIdx.x;
  if (tid < 128) sv[tid] = s[b*128 + tid];
  __syncthreads();
  if (tid < 128){
    float acc = fb1[tid];
    for (int k = 0; k < 128; ++k) acc += sv[k] * F1[k*128 + tid];
    tv[tid] = lrelu(acc);
  }
  __syncthreads();
  float acc = fb2[tid];
  for (int k = 0; k < 128; ++k) acc += tv[k] * F2[(size_t)k*512 + tid];
  out[b*512 + tid] = lrelu(acc);
}

// ---------------------------------------------------------------- launch
extern "C" void kernel_launch(void* const* d_in, const int* in_sizes, int n_in,
                              void* d_out, int out_size, void* d_ws, size_t ws_size,
                              hipStream_t stream){
  (void)in_sizes; (void)n_in; (void)out_size; (void)ws_size;
  const float* x  = (const float*)d_in[0];
  const float* h  = (const float*)d_in[1];
  const float* cw[4] = {(const float*)d_in[2],(const float*)d_in[6],(const float*)d_in[10],(const float*)d_in[14]};
  const float* cb[4] = {(const float*)d_in[3],(const float*)d_in[7],(const float*)d_in[11],(const float*)d_in[15]};
  const float* gg[4] = {(const float*)d_in[4],(const float*)d_in[8],(const float*)d_in[12],(const float*)d_in[16]};
  const float* be[4] = {(const float*)d_in[5],(const float*)d_in[9],(const float*)d_in[13],(const float*)d_in[17]};
  const float* W1 = (const float*)d_in[18]; const float* b1  = (const float*)d_in[19];
  const float* W2 = (const float*)d_in[20]; const float* b2  = (const float*)d_in[21];
  const float* W3 = (const float*)d_in[22]; const float* b3  = (const float*)d_in[23];
  const float* W4 = (const float*)d_in[24]; const float* b4  = (const float*)d_in[25];
  const float* F1 = (const float*)d_in[26]; const float* fb1 = (const float*)d_in[27];
  const float* F2 = (const float*)d_in[28]; const float* fb2 = (const float*)d_in[29];
  float* out = (float*)d_out;

  char* ws = (char*)d_ws;
  size_t off = 0;
  auto alloc = [&](size_t bytes)->char*{
    off = (off + 511) & ~(size_t)511;
    char* p = ws + off; off += bytes; return p;
  };
  float* s_acc = (float*)alloc(4096u*4);
  float* stats = (float*)alloc(2048u*4);                 // 4 layers x (sum256, sumsq256)
  float* act   = (float*)alloc((size_t)819200*4);        // (B,256,L) f32
  unsigned short* in0  = (unsigned short*)alloc((size_t)1638400*2);  // (B,512,L) bf16
  unsigned short* bufA = (unsigned short*)alloc((size_t)819200*2);
  unsigned short* bufB = (unsigned short*)alloc((size_t)819200*2);
  unsigned short* wcf[4];
  wcf[0] = (unsigned short*)alloc((size_t)256*1536*2);
  wcf[1] = (unsigned short*)alloc((size_t)256*768*2);
  wcf[2] = (unsigned short*)alloc((size_t)256*768*2);
  wcf[3] = (unsigned short*)alloc((size_t)256*768*2);
  unsigned short* wab  = (unsigned short*)alloc((size_t)2*9*8*512*2);
  unsigned short* wrel = (unsigned short*)alloc((size_t)3*4*8*512*2);
  unsigned short* xf   = (unsigned short*)alloc((size_t)3200*288*2);
  float* Abuf = (float*)alloc((size_t)3216*128*4);       // 16 pad rows for masked j-tiles
  float* Cbuf = (float*)alloc((size_t)3216*128*4);
  float* hb   = (float*)alloc(4096u*4);

  // zero accumulators
  k_zero<<<16, 256, 0, stream>>>(s_acc, 4096);
  k_zero<<<8, 256, 0, stream>>>(stats, 2048);

  // pack operands
  k_prep_in0<<<(1638400+255)/256, 256, 0, stream>>>(x, in0);
  k_prep_afrag<<<(16*48*512+255)/256, 256, 0, stream>>>(cw[0], wcf[0], 48);
  for (int l = 1; l < 4; ++l)
    k_prep_afrag<<<(16*24*512+255)/256, 256, 0, stream>>>(cw[l], wcf[l], 24);
  k_prep_bfrag<<<(9*8*512+255)/256, 256, 0, stream>>>(W1,              wab,            9, 258); // Wa
  k_prep_bfrag<<<(9*8*512+255)/256, 256, 0, stream>>>(W1 + 258*128,    wab + 9*8*512,  9, 258); // Wb
  k_prep_bfrag<<<(4*8*512+255)/256, 256, 0, stream>>>(W2, wrel,          4, 128);
  k_prep_bfrag<<<(4*8*512+255)/256, 256, 0, stream>>>(W3, wrel + 16384,  4, 128);
  k_prep_bfrag<<<(4*8*512+255)/256, 256, 0, stream>>>(W4, wrel + 32768,  4, 128);

  // conv stack
  const unsigned short* cin = in0; int Cin = 512;
  unsigned short* outs[4] = {bufA, bufB, bufA, bufB};
  for (int layer = 0; layer < 4; ++layer){
    k_conv_wmma<<<dim3(32,7,4), 128, 0, stream>>>(cin, wcf[layer], cb[layer], act,
                                                  stats + layer*512, Cin);
    k_bn<<<(819200+255)/256, 256, 0, stream>>>(act, stats + layer*512, gg[layer], be[layer],
                                               outs[layer]);
    cin = outs[layer]; Cin = 256;
  }

  // relation features
  k_prep_xf<<<(3200*288+255)/256, 256, 0, stream>>>(outs[3], xf);
  k_hb<<<16, 256, 0, stream>>>(h, W1, b1, hb);
  k_gemm_xf<<<200, 256, 0, stream>>>(xf, wab,            nullptr, Abuf);  // A = xf@Wa
  k_gemm_xf<<<200, 256, 0, stream>>>(xf, wab + 9*8*512,  hb,      Cbuf);  // C = xf@Wb + hb

  // fused pairwise MLP + reduction (dominant phase): 2 i-rows per weight fetch
  k_pairwise<<<dim3(32,13), 256, 0, stream>>>(Abuf, Cbuf, wrel, b2, b3, b4, s_acc);

  // head
  k_head<<<32, 512, 0, stream>>>(s_acc, F1, fb1, F2, fb2, out);
}
